// NeuralMemory_44324062495168
// MI455X (gfx1250) — matrix operand order repaired and checked by reading, not compiled, not run on hardware
//
#include <hip/hip_runtime.h>
#include <hip/hip_bf16.h>
#include <math.h>

#define BB 8
#define SS 512
#define DD 512
#define HH 1024
#define CC 8
#define NCHUNK 64

typedef __attribute__((ext_vector_type(16))) _Float16 v16h;
typedef __attribute__((ext_vector_type(8)))  float    v8f;

__device__ __forceinline__ float sigmoidf_(float x) { return 1.0f / (1.0f + __expf(-x)); }
__device__ __forceinline__ float siluf_(float x)    { return x * sigmoidf_(x); }

// ---------------------------------------------------------------------------
// Phase 1a: Y = X @ W^T for Wk/Wv/Wq, via v_wmma_f32_16x16x32_f16.
// A layout (16-bit A 16x32): lane m=l%16; elems 0..7 -> K=koffA..+7, 8..15 -> K=16+koffA..+7, koffA=(l>=16)?8:0
// B layout (16-bit B 32x16): lane n=l%16; same K order as A (elems 0..7 -> koffA..+7, 8..15 -> 16+koffA..+7)
// C/D: VGPR v: lanes0-15 M=v, lanes16-31 M=v+8.
// ---------------------------------------------------------------------------
__global__ void proj_wmma_kernel(const float* __restrict__ x,
                                 const float* __restrict__ Wk,
                                 const float* __restrict__ Wv,
                                 const float* __restrict__ Wq,
                                 float* __restrict__ hk, float* __restrict__ hv,
                                 float* __restrict__ hq)
{
    const int lane = threadIdx.x & 31;
    const int wave = threadIdx.x >> 5;
    const int r0 = blockIdx.x * 16;
    const int n0 = (blockIdx.y * 4 + wave) * 16;
    const float* W; float* Y;
    if (blockIdx.z == 0)      { W = Wk; Y = hk; }
    else if (blockIdx.z == 1) { W = Wv; Y = hv; }
    else                      { W = Wq; Y = hq; }
    const int mrow  = lane & 15;
    const int koffA = (lane >> 4) * 8;
    const int koffB = (lane >> 4) * 16;
    const float* xrow = x + (size_t)(r0 + mrow) * DD;
    const float* wrow = W + (size_t)(n0 + mrow) * DD;
    v8f acc = {};
    for (int k0 = 0; k0 < DD; k0 += 32) {
        v16h a, bf;
#pragma unroll
        for (int j = 0; j < 8; ++j) {
            a[j]     = (_Float16)xrow[k0 + koffA + j];
            a[8 + j] = (_Float16)xrow[k0 + koffA + 16 + j];
        }
#pragma unroll
        for (int j = 0; j < 8; ++j) { bf[j] = (_Float16)wrow[k0 + koffA + j]; bf[8 + j] = (_Float16)wrow[k0 + koffA + 16 + j]; }
        acc = __builtin_amdgcn_wmma_f32_16x16x32_f16(false, a, false, bf,
                                                     (short)0, acc, false, false);
    }
    const int rbase = r0 + ((lane >> 4) << 3);
    const int ncol  = n0 + mrow;
#pragma unroll
    for (int v = 0; v < 8; ++v)
        Y[(size_t)(rbase + v) * DD + ncol] = acc[v];
}

// ---------------------------------------------------------------------------
// Phase 1b: causal conv(4) + bias + SiLU (+ L2 norm for k/q). One block per (b,s,tensor).
// ---------------------------------------------------------------------------
__global__ void conv_silu_norm_kernel(const float* __restrict__ hk, const float* __restrict__ hv,
                                      const float* __restrict__ hq,
                                      const float* __restrict__ ckw, const float* __restrict__ ckb,
                                      const float* __restrict__ cvw, const float* __restrict__ cvb,
                                      const float* __restrict__ cqw, const float* __restrict__ cqb,
                                      float* __restrict__ Kp, float* __restrict__ Vp,
                                      float* __restrict__ Qp)
{
    const int s = blockIdx.x, b = blockIdx.y, which = blockIdx.z;
    const float* h; const float* cw; const float* cb; float* out; bool donorm;
    if (which == 0)      { h = hk; cw = ckw; cb = ckb; out = Kp; donorm = true;  }
    else if (which == 1) { h = hv; cw = cvw; cb = cvb; out = Vp; donorm = false; }
    else                 { h = hq; cw = cqw; cb = cqb; out = Qp; donorm = true;  }
    const int tx = threadIdx.x;
    __shared__ float red[256];
    float y[2]; float ssum = 0.f;
#pragma unroll
    for (int e = 0; e < 2; ++e) {
        int d = tx + e * 256;
        float acc = cb[d];
#pragma unroll
        for (int j = 0; j < 4; ++j) {
            int si = s - 3 + j;
            if (si >= 0) acc += h[((size_t)b * SS + si) * DD + d] * cw[d * 4 + j];
        }
        float v = siluf_(acc);
        y[e] = v; ssum += v * v;
    }
    size_t rowoff = ((size_t)b * SS + s) * DD;
    if (donorm) {
        red[tx] = ssum; __syncthreads();
        for (int o = 128; o > 0; o >>= 1) { if (tx < o) red[tx] += red[tx + o]; __syncthreads(); }
        float scale = 1.0f / fmaxf(sqrtf(red[0]), 1e-12f);
        out[rowoff + tx]       = y[0] * scale;
        out[rowoff + tx + 256] = y[1] * scale;
    } else {
        out[rowoff + tx]       = y[0];
        out[rowoff + tx + 256] = y[1];
    }
}

// ---------------------------------------------------------------------------
// Phase 1c: theta/eta/alpha gates. One block per (b,s).
// ---------------------------------------------------------------------------
__global__ void gates_kernel(const float* __restrict__ x,
                             const float* __restrict__ thw, const float* __restrict__ thb,
                             const float* __restrict__ etw, const float* __restrict__ etb,
                             const float* __restrict__ alw, const float* __restrict__ alb,
                             float* __restrict__ theta, float* __restrict__ eta,
                             float* __restrict__ alpha)
{
    const int s = blockIdx.x, b = blockIdx.y, tx = threadIdx.x;
    __shared__ float r0[256], r1[256], r2[256];
    const float* xr = x + ((size_t)b * SS + s) * DD;
    float a0 = 0.f, a1 = 0.f, a2 = 0.f;
    for (int d = tx; d < DD; d += 256) {
        float xv = xr[d];
        a0 += xv * thw[d]; a1 += xv * etw[d]; a2 += xv * alw[d];
    }
    r0[tx] = a0; r1[tx] = a1; r2[tx] = a2; __syncthreads();
    for (int o = 128; o > 0; o >>= 1) {
        if (tx < o) { r0[tx] += r0[tx+o]; r1[tx] += r1[tx+o]; r2[tx] += r2[tx+o]; }
        __syncthreads();
    }
    if (tx == 0) {
        theta[b * SS + s] = sigmoidf_(r0[0] + thb[0]);
        eta[b * SS + s]   = sigmoidf_(r1[0] + etb[0]);
        alpha[b * SS + s] = sigmoidf_(r2[0] + alb[0]);
    }
}

// ---------------------------------------------------------------------------
// State init: W0=M0 bcast, W1=M1 bcast, m0=m1=0.
// ---------------------------------------------------------------------------
__global__ void init_state_kernel(const float* __restrict__ M0, const float* __restrict__ M1,
                                  float* __restrict__ W0, float* __restrict__ W1,
                                  float* __restrict__ m0, float* __restrict__ m1)
{
    size_t i = (size_t)blockIdx.x * blockDim.x + threadIdx.x;
    size_t n = (size_t)BB * HH * DD;
    if (i >= n) return;
    size_t j = i & ((size_t)HH * DD - 1);
    W0[i] = M0[j]; W1[i] = M1[j]; m0[i] = 0.f; m1[i] = 0.f;
}

// z for token 0 -> a(z), sigterm(z). One wave per (b,h).
__global__ void zinit_kernel(const float* __restrict__ M0, const float* __restrict__ Kp,
                             float* __restrict__ a_buf, float* __restrict__ st_buf)
{
    int w    = (int)((blockIdx.x * blockDim.x + threadIdx.x) >> 5);
    int lane = threadIdx.x & 31;
    int b = w >> 10, h = w & (HH - 1);
    const float* m0r = M0 + (size_t)h * DD;
    const float* kr  = Kp + (size_t)b * SS * DD;   // token 0
    float acc = 0.f;
    for (int d = lane; d < DD; d += 32) acc += m0r[d] * kr[d];
    for (int o = 16; o > 0; o >>= 1) acc += __shfl_xor(acc, o);
    if (lane == 0) {
        float sig = sigmoidf_(acc);
        a_buf[b * HH + h]  = acc * sig;
        st_buf[b * HH + h] = sig * (1.0f + acc * (1.0f - sig));
    }
}

// ---------------------------------------------------------------------------
// Chunk output stage 1: zs[b,i,h] = silu(sum_d q[b,c*8+i,d]*W0[b,h,d]), M padded 8->16 (WMMA).
// ---------------------------------------------------------------------------
__global__ void chunk_zs_kernel(const float* __restrict__ Qp, const float* __restrict__ W0,
                                float* __restrict__ zs, int c)
{
    const int lane = threadIdx.x & 31;
    const int wave = threadIdx.x >> 5;
    const int b  = blockIdx.y;
    const int n0 = (blockIdx.x * 4 + wave) * 16;
    const int mrow  = lane & 15;
    const int koffA = (lane >> 4) * 8;
    const int koffB = (lane >> 4) * 16;
    const bool avalid = mrow < CC;
    const float* qrow  = Qp + ((size_t)b * SS + (c * CC + (avalid ? mrow : 0))) * DD;
    const float* w0row = W0 + ((size_t)b * HH + n0 + mrow) * DD;
    v8f acc = {};
    for (int k0 = 0; k0 < DD; k0 += 32) {
        v16h a, bf;
#pragma unroll
        for (int j = 0; j < 8; ++j) {
            a[j]     = avalid ? (_Float16)qrow[k0 + koffA + j]      : (_Float16)0.f;
            a[8 + j] = avalid ? (_Float16)qrow[k0 + koffA + 16 + j] : (_Float16)0.f;
        }
#pragma unroll
        for (int j = 0; j < 8; ++j) { bf[j] = (_Float16)w0row[k0 + koffA + j]; bf[8 + j] = (_Float16)w0row[k0 + koffA + 16 + j]; }
        acc = __builtin_amdgcn_wmma_f32_16x16x32_f16(false, a, false, bf,
                                                     (short)0, acc, false, false);
    }
    if (lane < 16) {          // lanes 0..15 hold M=0..7 in acc[v]
#pragma unroll
        for (int v = 0; v < 8; ++v)
            zs[((size_t)b * CC + v) * HH + n0 + mrow] = siluf_(acc[v]);
    }
}

// Chunk output stage 2: out[b,c*8+i,d] = sum_h zs[b,i,h]*W1[b,d,h] (WMMA).
__global__ void chunk_out_kernel(const float* __restrict__ zs, const float* __restrict__ W1,
                                 float* __restrict__ outp, int c)
{
    const int lane = threadIdx.x & 31;
    const int wave = threadIdx.x >> 5;
    const int b  = blockIdx.y;
    const int n0 = (blockIdx.x * 4 + wave) * 16;      // d tile
    const int mrow  = lane & 15;
    const int koffA = (lane >> 4) * 8;
    const int koffB = (lane >> 4) * 16;
    const bool avalid = mrow < CC;
    const float* arow  = zs + ((size_t)b * CC + (avalid ? mrow : 0)) * HH;
    const float* w1row = W1 + ((size_t)b * DD + n0 + mrow) * HH;
    v8f acc = {};
    for (int k0 = 0; k0 < HH; k0 += 32) {
        v16h a, bf;
#pragma unroll
        for (int j = 0; j < 8; ++j) {
            a[j]     = avalid ? (_Float16)arow[k0 + koffA + j]      : (_Float16)0.f;
            a[8 + j] = avalid ? (_Float16)arow[k0 + koffA + 16 + j] : (_Float16)0.f;
        }
#pragma unroll
        for (int j = 0; j < 8; ++j) { bf[j] = (_Float16)w1row[k0 + koffA + j]; bf[8 + j] = (_Float16)w1row[k0 + koffA + 16 + j]; }
        acc = __builtin_amdgcn_wmma_f32_16x16x32_f16(false, a, false, bf,
                                                     (short)0, acc, false, false);
    }
    if (lane < 16) {
#pragma unroll
        for (int v = 0; v < 8; ++v)
            outp[((size_t)b * SS + c * CC + v) * DD + n0 + mrow] = acc[v];
    }
}

// ---------------------------------------------------------------------------
// Token step 1: dpred[b,d] = sum_h a[b,h]*W1[b,d,h] - v[b,t,d]. One wave per (b,d).
// All 8 waves of a block share batch b -> stage a_buf row (4 KB) in LDS once.
// ---------------------------------------------------------------------------
__global__ void token_pred_kernel(const float* __restrict__ a_buf, const float* __restrict__ W1,
                                  const float* __restrict__ Vp, float* __restrict__ dpred, int t)
{
    __shared__ float sa[HH];
    const int wave = threadIdx.x >> 5;
    const int lane = threadIdx.x & 31;
    const int w = blockIdx.x * 8 + wave;
    const int b = w >> 9;             // DD = 512; 8 waves/block never straddle a batch
    const int d = w & (DD - 1);
    for (int h = threadIdx.x; h < HH; h += 256)
        sa[h] = a_buf[(size_t)b * HH + h];
    __syncthreads();
    const float* w1 = W1 + ((size_t)b * DD + d) * HH;
    float acc = 0.f;
    for (int h = lane; h < HH; h += 32) acc += sa[h] * w1[h];
    for (int o = 16; o > 0; o >>= 1) acc += __shfl_xor(acc, o);
    if (lane == 0)
        dpred[b * DD + d] = acc - Vp[((size_t)b * SS + t) * DD + d];
}

// Token step 2 (fused): dz reduction over old W1 + m1/W1 update, single pass.
// Block = 32 h-lanes x 8 d-groups; coalesced across h for fixed d.
// dpred (2 KB) staged in LDS; W1/m1 stream prefetched ~96 KB ahead
// (4 KB-strided walk that HW prefetch won't catch; state is L2-resident).
__global__ void token_w1_kernel(const float* __restrict__ dpred, const float* __restrict__ a_buf,
                                const float* __restrict__ st_buf,
                                const float* __restrict__ theta, const float* __restrict__ eta,
                                const float* __restrict__ alpha,
                                float* __restrict__ W1, float* __restrict__ m1,
                                float* __restrict__ dz, int t)
{
    __shared__ float sdp[DD];
    const int tid = threadIdx.x;
    const int b   = blockIdx.y;
    for (int d = tid; d < DD; d += 256) sdp[d] = dpred[b * DD + d];
    __syncthreads();
    const int tx = tid & 31;   // h lane
    const int ty = tid >> 5;   // d group
    const int h  = blockIdx.x * 32 + tx;
    const float th = theta[b * SS + t];
    const float et = eta[b * SS + t];
    const float al = alpha[b * SS + t];
    const float ah = a_buf[b * HH + h];
    float dzp = 0.f;
    const size_t base = (size_t)b * DD * HH + h;
    for (int d0 = ty * 64; d0 < ty * 64 + 64; d0 += 8) {
        __builtin_prefetch(&W1[base + (size_t)(d0 + 24) * HH], 0, 1);
        __builtin_prefetch(&m1[base + (size_t)(d0 + 24) * HH], 0, 1);
#pragma unroll
        for (int d = d0; d < d0 + 8; ++d) {
            size_t idx = base + (size_t)d * HH;
            float w  = W1[idx];                  // old W1 for dz
            float dp = sdp[d];
            dzp += dp * w;
            float m1n = et * m1[idx] - th * dp * ah;
            m1[idx] = m1n;
            W1[idx] = (1.0f - al) * w + m1n;
        }
    }
    __shared__ float red[8][32];
    red[ty][tx] = dzp;
    __syncthreads();
    if (ty == 0) {
        float s = 0.f;
#pragma unroll
        for (int j = 0; j < 8; ++j) s += red[j][tx];
        dz[b * HH + h] = s * st_buf[b * HH + h];
    }
}

// Token step 3 (fused): m0/W0 update + z(t+1) matvec with the fresh W0 row,
// then a(z), sigterm(z) for the next token. One block per (b,h).
__global__ void token_w0_kernel(const float* __restrict__ dz, const float* __restrict__ Kp,
                                const float* __restrict__ theta, const float* __restrict__ eta,
                                const float* __restrict__ alpha,
                                float* __restrict__ W0, float* __restrict__ m0,
                                float* __restrict__ a_buf, float* __restrict__ st_buf, int t)
{
    const int h = blockIdx.x;
    const int b = blockIdx.y;
    const int tx = threadIdx.x;
    const float th  = theta[b * SS + t];
    const float et  = eta[b * SS + t];
    const float al  = alpha[b * SS + t];
    const float dzh = dz[b * HH + h];
    const int t1 = (t + 1 < SS) ? t + 1 : t;
    const float* kt = Kp + ((size_t)b * SS + t)  * DD;
    const float* kn = Kp + ((size_t)b * SS + t1) * DD;
    const size_t base = ((size_t)b * HH + h) * DD;
    float zacc = 0.f;
#pragma unroll
    for (int e = 0; e < 2; ++e) {
        int d = tx + e * 256;
        size_t idx = base + d;
        float m0n = et * m0[idx] - th * dzh * kt[d];
        m0[idx] = m0n;
        float w0n = (1.0f - al) * W0[idx] + m0n;
        W0[idx] = w0n;
        zacc += w0n * kn[d];
    }
    __shared__ float red[256];
    red[tx] = zacc; __syncthreads();
    for (int o = 128; o > 0; o >>= 1) { if (tx < o) red[tx] += red[tx + o]; __syncthreads(); }
    if (tx == 0) {
        float z = red[0];
        float sig = sigmoidf_(z);
        a_buf[b * HH + h]  = z * sig;
        st_buf[b * HH + h] = sig * (1.0f + z * (1.0f - sig));
    }
}

// Final LayerNorm, in place over d_out. One block per (b,s).
__global__ void layernorm_kernel(float* __restrict__ out, const float* __restrict__ g,
                                 const float* __restrict__ bta)
{
    const int s = blockIdx.x, b = blockIdx.y, tx = threadIdx.x;
    float* row = out + ((size_t)b * SS + s) * DD;
    __shared__ float r0[256], r1[256];
    float v0 = row[tx], v1 = row[tx + 256];
    r0[tx] = v0 + v1;
    r1[tx] = v0 * v0 + v1 * v1;
    __syncthreads();
    for (int o = 128; o > 0; o >>= 1) {
        if (tx < o) { r0[tx] += r0[tx + o]; r1[tx] += r1[tx + o]; }
        __syncthreads();
    }
    float mu  = r0[0] * (1.0f / DD);
    float var = r1[0] * (1.0f / DD) - mu * mu;
    float inv = rsqrtf(var + 1e-5f);
    row[tx]       = (v0 - mu) * inv * g[tx]       + bta[tx];
    row[tx + 256] = (v1 - mu) * inv * g[tx + 256] + bta[tx + 256];
}

extern "C" void kernel_launch(void* const* d_in, const int* in_sizes, int n_in,
                              void* d_out, int out_size, void* d_ws, size_t ws_size,
                              hipStream_t stream)
{
    const float* x   = (const float*)d_in[0];
    const float* Wk  = (const float*)d_in[1];
    const float* Wv  = (const float*)d_in[2];
    const float* Wq  = (const float*)d_in[3];
    const float* ckw = (const float*)d_in[4];
    const float* ckb = (const float*)d_in[5];
    const float* cvw = (const float*)d_in[6];
    const float* cvb = (const float*)d_in[7];
    const float* cqw = (const float*)d_in[8];
    const float* cqb = (const float*)d_in[9];
    const float* M0  = (const float*)d_in[10];
    const float* M1  = (const float*)d_in[11];
    const float* thw = (const float*)d_in[12];
    const float* thb = (const float*)d_in[13];
    const float* etw = (const float*)d_in[14];
    const float* etb = (const float*)d_in[15];
    const float* alw = (const float*)d_in[16];
    const float* alb = (const float*)d_in[17];
    const float* lng = (const float*)d_in[18];
    const float* lnb = (const float*)d_in[19];
    float* outp = (float*)d_out;
    float* ws   = (float*)d_ws;

    const size_t BSD = (size_t)BB * SS * DD;   // 2,097,152
    const size_t BHD = (size_t)BB * HH * DD;   // 4,194,304
    float* hk = ws;              float* hv = hk + BSD;    float* hq = hv + BSD;
    float* Kp = hq + BSD;        float* Vp = Kp + BSD;    float* Qp = Vp + BSD;
    float* W0 = Qp + BSD;        float* W1 = W0 + BHD;
    float* m0 = W1 + BHD;        float* m1 = m0 + BHD;
    float* theta = m1 + BHD;     float* eta = theta + (size_t)BB * SS;
    float* alpha = eta + (size_t)BB * SS;
    float* zs     = alpha + (size_t)BB * SS;
    float* a_buf  = zs + (size_t)BB * CC * HH;
    float* st_buf = a_buf + (size_t)BB * HH;
    float* dpred  = st_buf + (size_t)BB * HH;
    float* dz     = dpred + (size_t)BB * DD;

    // Phase 1
    proj_wmma_kernel<<<dim3(BB * SS / 16, DD / 64, 3), 128, 0, stream>>>(
        x, Wk, Wv, Wq, hk, hv, hq);
    conv_silu_norm_kernel<<<dim3(SS, BB, 3), 256, 0, stream>>>(
        hk, hv, hq, ckw, ckb, cvw, cvb, cqw, cqb, Kp, Vp, Qp);
    gates_kernel<<<dim3(SS, BB), 256, 0, stream>>>(
        x, thw, thb, etw, etb, alw, alb, theta, eta, alpha);
    init_state_kernel<<<(unsigned)((BHD + 255) / 256), 256, 0, stream>>>(
        M0, M1, W0, W1, m0, m1);
    zinit_kernel<<<BB * HH / 8, 256, 0, stream>>>(M0, Kp, a_buf, st_buf);

    // Phase 2: chunked sequential scan
    for (int c = 0; c < NCHUNK; ++c) {
        chunk_zs_kernel<<<dim3(HH / 64, BB), 128, 0, stream>>>(Qp, W0, zs, c);
        chunk_out_kernel<<<dim3(DD / 64, BB), 128, 0, stream>>>(zs, W1, outp, c);
        for (int i = 0; i < CC; ++i) {
            int t = c * CC + i;
            token_pred_kernel<<<BB * DD / 8, 256, 0, stream>>>(a_buf, W1, Vp, dpred, t);
            token_w1_kernel<<<dim3(HH / 32, BB), 256, 0, stream>>>(
                dpred, a_buf, st_buf, theta, eta, alpha, W1, m1, dz, t);
            token_w0_kernel<<<dim3(HH, BB), 256, 0, stream>>>(
                dz, Kp, theta, eta, alpha, W0, m0, a_buf, st_buf, t);
        }
    }
    layernorm_kernel<<<dim3(SS, BB), 256, 0, stream>>>(outp, lng, lnb);
}
